// MultiHeadedAttention_spatial_36575941492790
// MI455X (gfx1250) — matrix operand order repaired and checked by reading, not compiled, not run on hardware
//
#include <hip/hip_runtime.h>
#include <hip/hip_bf16.h>

typedef __attribute__((ext_vector_type(16))) _Float16 v16h;
typedef __attribute__((ext_vector_type(8)))  float    v8f;

union AFrag { v16h v; uint4 q[2]; };

// ---------------------------------------------------------------------------
// One-time fp32 -> f16 convert. Thread t converts 8 contiguous elements:
// 2x global_load_b128 in, 1x global_store_b128 out. n must be multiple of 8.
// ---------------------------------------------------------------------------
__global__ __launch_bounds__(256)
void cvt_f32_to_f16(const float* __restrict__ in, _Float16* __restrict__ out, int n)
{
    int base = (blockIdx.x * 256 + threadIdx.x) * 8;
    if (base >= n) return;
    #pragma unroll
    for (int i = 0; i < 8; ++i)
        out[base + i] = (_Float16)in[base + i];
}

// ---------------------------------------------------------------------------
// GEMM (NT):  out[M x 256] = A16[M x 256] @ W16[256 x 256]^T + bias
// A16, W16 f16 in global; fp32 accumulate via v_wmma_f32_16x16x32_f16.
// Block = 128 threads = 4 waves; block computes 16 rows x 256 cols.
// A tile staged in LDS once; its 8 K-step fragments hoisted to registers and
// reused across 4 column tiles per wave. B fragment = 2x global_load_b128.
// ---------------------------------------------------------------------------
__global__ __launch_bounds__(128)
void gemm_nt_256_f16(const _Float16* __restrict__ A16,
                     const _Float16* __restrict__ W16,
                     const float* __restrict__ bias,
                     float* __restrict__ out)
{
    __shared__ _Float16 ldsA[16 * 256];   // 8 KB f16 A tile

    const int tid     = threadIdx.x;
    const int rowBase = blockIdx.x * 16;
    const int waveId  = tid >> 5;
    const int lane    = tid & 31;
    const int half    = lane >> 4;        // 0: lanes 0-15, 1: lanes 16-31
    const int m       = lane & 15;

    // Stage A tile: 4096 halves = 512 uint4, 128 threads x 4 vec loads.
    {
        const uint4* src = (const uint4*)(A16 + (size_t)rowBase * 256);
        uint4* dst = (uint4*)ldsA;
        #pragma unroll
        for (int c = 0; c < 4; ++c)
            dst[tid + c * 128] = src[tid + c * 128];
    }
    __syncthreads();

    // A 16x32 f16 fragment layout (wave32): lane row M = lane&15;
    // VGPRs 0-3 hold K = 0..7 (+8*half), VGPRs 4-7 hold K = 16..23 (+8*half)
    // -> two aligned 16-byte LDS chunks per fragment.
    AFrag afrag[8];
    #pragma unroll
    for (int ks = 0; ks < 8; ++ks) {
        const _Float16* ap = ldsA + m * 256 + ks * 32 + 8 * half;
        afrag[ks].q[0] = *(const uint4*)(ap);
        afrag[ks].q[1] = *(const uint4*)(ap + 16);
    }

    const int n = lane & 15;
    #pragma unroll
    for (int t = 0; t < 4; ++t) {
        const int colBase = (waveId * 4 + t) * 16;
        const int wrow    = colBase + n;              // W row == output column
        const _Float16* wp = W16 + (size_t)wrow * 256;

        v8f c = {0.f, 0.f, 0.f, 0.f, 0.f, 0.f, 0.f, 0.f};

        #pragma unroll
        for (int ks = 0; ks < 8; ++ks) {
            // B 32x16 f16 layout: lane column n = lane&15; VGPRs 0-3 hold
            // K = 0..7 (+8*half), VGPRs 4-7 hold K = 16..23 (+8*half).
            AFrag bf;
            const _Float16* bp = wp + ks * 32 + 8 * half;
            bf.q[0] = *(const uint4*)(bp);
            bf.q[1] = *(const uint4*)(bp + 16);
            c = __builtin_amdgcn_wmma_f32_16x16x32_f16(
                    false, afrag[ks].v, false, bf.v,
                    (short)0, c, false, false);
        }

        const float bv = bias[colBase + n];
        #pragma unroll
        for (int r = 0; r < 8; ++r) {
            int row = rowBase + 8 * half + r;          // C: VGPR r -> row 8*half+r
            out[(size_t)row * 256 + colBase + n] = c[r] + bv;
        }
    }
}

// ---------------------------------------------------------------------------
// Gather attention: one block per node (256 thr = 8 waves), one wave per head.
// Lane j owns neighbor j: score -> mask -> wave32-shuffle softmax; p/index
// bounce through LDS so lane d owns output dim d (coalesced v reads).
// k/v tables (16 MB each) stay L2-resident (192 MB L2). Writes x in f16 so
// the output-projection GEMM consumes it directly.
// ---------------------------------------------------------------------------
__global__ __launch_bounds__(256)
void attn_gather(const float* __restrict__ qb,
                 const float* __restrict__ kb,
                 const float* __restrict__ vb,
                 const int*   __restrict__ nbrs,
                 const int*   __restrict__ mask,
                 _Float16* __restrict__ x16)
{
    __shared__ float sp[8][32];
    __shared__ int   snb[8][32];

    const int b    = blockIdx.x;
    const int tid  = threadIdx.x;
    const int h    = tid >> 5;        // head
    const int lane = tid & 31;        // neighbor slot / output dim

    const int nj = nbrs[(size_t)b * 32 + lane];
    const int mj = mask[(size_t)b * 32 + lane];

    const float* qrow = qb + (size_t)b  * 256 + h * 32;   // wave-uniform -> SMEM
    const float* krow = kb + (size_t)nj * 256 + h * 32;

    float s = 0.f;
    #pragma unroll
    for (int d = 0; d < 32; ++d) s += qrow[d] * krow[d];
    s *= 0.17677669529663689f;                 // 1/sqrt(32)
    s = (mj == 0) ? -1.0e10f : s;

    float mx = s;
    #pragma unroll
    for (int off = 16; off >= 1; off >>= 1)
        mx = fmaxf(mx, __shfl_xor(mx, off, 32));
    float e = __expf(s - mx);
    float sum = e;
    #pragma unroll
    for (int off = 16; off >= 1; off >>= 1)
        sum += __shfl_xor(sum, off, 32);
    const float p = e / sum;

    sp[h][lane]  = p;
    snb[h][lane] = nj;
    __syncthreads();

    float o = 0.f;
    #pragma unroll 8
    for (int j = 0; j < 32; ++j)
        o += sp[h][j] * vb[(size_t)snb[h][j] * 256 + h * 32 + lane];

    x16[(size_t)b * 256 + h * 32 + lane] = (_Float16)o;
}

// ---------------------------------------------------------------------------
// Harness entry. Inputs (setup_inputs order):
//  0 query_(NB,256) f32 | 1 spatial_neighbors(NB,32) int | 2 mask(NB,1,32) int
//  3 Wq 4 bq 5 Wk 6 bk 7 Wv 8 bv 9 Wo 10 bo   (all f32)
// Output: (NB,1,256) f32.
// Workspace: q/k/v f32 (48 MB) + x16 (8 MB) + q16 (8 MB) + 4x W16 (0.5 MB).
// ---------------------------------------------------------------------------
extern "C" void kernel_launch(void* const* d_in, const int* in_sizes, int n_in,
                              void* d_out, int out_size, void* d_ws, size_t ws_size,
                              hipStream_t stream)
{
    const float* query = (const float*)d_in[0];
    const int*   nbrs  = (const int*)  d_in[1];
    const int*   mask  = (const int*)  d_in[2];
    const float* Wq    = (const float*)d_in[3];
    const float* bq    = (const float*)d_in[4];
    const float* Wk    = (const float*)d_in[5];
    const float* bk    = (const float*)d_in[6];
    const float* Wv    = (const float*)d_in[7];
    const float* bv    = (const float*)d_in[8];
    const float* Wo    = (const float*)d_in[9];
    const float* bo    = (const float*)d_in[10];

    const int nb    = in_sizes[0] / 256;       // 16384
    const int nElem = nb * 256;                // 4,194,304
    const int wElem = 256 * 256;               // 65,536

    // Workspace carve-up
    float*    qb   = (float*)d_ws;
    float*    kbuf = qb   + (size_t)nElem;
    float*    vbuf = kbuf + (size_t)nElem;
    _Float16* x16  = (_Float16*)(vbuf + (size_t)nElem);
    _Float16* q16  = x16 + (size_t)nElem;
    _Float16* wq16 = q16 + (size_t)nElem;
    _Float16* wk16 = wq16 + wElem;
    _Float16* wv16 = wk16 + wElem;
    _Float16* wo16 = wv16 + wElem;

    // One-time fp32 -> f16 conversions (amortized across all GEMM blocks).
    cvt_f32_to_f16<<<(nElem / 8 + 255) / 256, 256, 0, stream>>>(query, q16, nElem);
    cvt_f32_to_f16<<<(wElem / 8 + 255) / 256, 256, 0, stream>>>(Wq, wq16, wElem);
    cvt_f32_to_f16<<<(wElem / 8 + 255) / 256, 256, 0, stream>>>(Wk, wk16, wElem);
    cvt_f32_to_f16<<<(wElem / 8 + 255) / 256, 256, 0, stream>>>(Wv, wv16, wElem);
    cvt_f32_to_f16<<<(wElem / 8 + 255) / 256, 256, 0, stream>>>(Wo, wo16, wElem);

    const int gemmBlocks = nb / 16;

    gemm_nt_256_f16<<<gemmBlocks, 128, 0, stream>>>(q16, wq16, bq, qb);
    gemm_nt_256_f16<<<gemmBlocks, 128, 0, stream>>>(q16, wk16, bk, kbuf);
    gemm_nt_256_f16<<<gemmBlocks, 128, 0, stream>>>(q16, wv16, bv, vbuf);

    attn_gather<<<nb, 256, 0, stream>>>(qb, kbuf, vbuf, nbrs, mask, x16);

    gemm_nt_256_f16<<<gemmBlocks, 128, 0, stream>>>(x16, wo16, bo, (float*)d_out);
}
